// MultiHeadDifferentialAttention_8615704396329
// MI455X (gfx1250) — compile-verified, matrix operand order checked by reading, not run on hardware
//
#include <hip/hip_runtime.h>
#include <hip/hip_bf16.h>
#include <math.h>

typedef __attribute__((ext_vector_type(16))) _Float16 v16h;
typedef __attribute__((ext_vector_type(8)))  _Float16 h8;
typedef __attribute__((ext_vector_type(8)))  float    v8f;

#define D_MODEL 1024
#define HEADS   16
#define DH      64
#define DH2     128            // 2*DH per head
#define BATCH   2
#define SEQ     2048
#define TOKENS  (BATCH*SEQ)    // 4096
#define QKV_N   (2*D_MODEL)    // 2048
#define LAMBDA_INIT 0.8f
#define EPS_RMS 1e-5f

// ---- gfx1250 async global->LDS staging ------------------------------------
// The clang builtin exists but takes TDM-style v4i32 descriptor groups
// (signature uncharacterized), so we emit the VGLOBAL instruction directly.
// ISA 15.18.3 op 98: global_load_async_to_lds_b128 vdst(LDS addr), vaddr, off
// LDS address = low 32 bits of the flat pointer (ISA 10.2 aperture mapping).
// Tracked by ASYNCcnt -> s_wait_asynccnt.
#define USE_ASYNC_LDS 1

#if USE_ASYNC_LDS
__device__ __forceinline__ void stage_b128(const _Float16* g, _Float16* l) {
  unsigned lds_addr = (unsigned)(size_t)(void*)l;
  asm volatile("global_load_async_to_lds_b128 %0, %1, off"
               :: "v"(lds_addr), "v"(g) : "memory");
}
#define WAIT_ASYNC() asm volatile("s_wait_asynccnt 0x0" ::: "memory")
#else
__device__ __forceinline__ void stage_b128(const _Float16* g, _Float16* l) {
  *(h8*)l = *(const h8*)g;   // synchronous load + ds_store fallback
}
#define WAIT_ASYNC() ((void)0)
#endif

// ---------------------------------------------------------------------------
// Load a 16-bit WMMA A/B operand (K=32) for one lane.
// Layout (ISA 7.12.2, 16-bit A 16x32): lane (r = lane%16, half = lane/16)
// holds halves[0..7]  = row r, K = half*8 + 0..7
//       halves[8..15] = row r, K = 16 + half*8 + 0..7
// `p` = start of this lane's row (K-contiguous, 16B-aligned).
// ---------------------------------------------------------------------------
__device__ __forceinline__ v16h load_op(const _Float16* p, int halfsel) {
  h8 lo = *(const h8*)(p + halfsel * 8);
  h8 hi = *(const h8*)(p + 16 + halfsel * 8);
  v16h r;
#pragma unroll
  for (int i = 0; i < 8; ++i) { r[i] = lo[i]; r[i + 8] = hi[i]; }
  return r;
}

__device__ __forceinline__ v8f wmma32(v16h a, v16h b, v8f c) {
  // D = A(16x32) * B(32x16) + C, f32 accumulate
  return __builtin_amdgcn_wmma_f32_16x16x32_f16(
      /*neg_a=*/false, a, /*neg_b=*/false, b,
      /*c_mod=*/(short)0, c, /*reuse_a=*/false, /*reuse_b=*/false);
}

// ---------------------------------------------------------------------------
// Elementwise f32 -> f16
// ---------------------------------------------------------------------------
__global__ void k_f32_to_f16(const float* __restrict__ src,
                             _Float16* __restrict__ dst, int n) {
  int i = blockIdx.x * blockDim.x + threadIdx.x;
  int stride = gridDim.x * blockDim.x;
  for (; i < n; i += stride) dst[i] = (_Float16)src[i];
}

// Transpose src[rows][cols] (f32) -> dst[cols][rows] (f16)
__global__ void k_transpose_f16(const float* __restrict__ src,
                                _Float16* __restrict__ dst, int rows, int cols) {
  int i = blockIdx.x * blockDim.x + threadIdx.x;
  int stride = gridDim.x * blockDim.x;
  int n = rows * cols;
  for (; i < n; i += stride) {
    int c = i / rows, r = i % rows;
    dst[i] = (_Float16)src[(size_t)r * cols + c];
  }
}

// lam[h] = exp(lq1.lk1) - exp(lq2.lk2) + lambda_init
__global__ void k_lambda(const float* __restrict__ lq1, const float* __restrict__ lk1,
                         const float* __restrict__ lq2, const float* __restrict__ lk2,
                         float* __restrict__ lam) {
  int h = threadIdx.x;
  if (h < HEADS) {
    float s1 = 0.f, s2 = 0.f;
    for (int i = 0; i < DH; ++i) {
      s1 += lq1[h * DH + i] * lk1[h * DH + i];
      s2 += lq2[h * DH + i] * lk2[h * DH + i];
    }
    lam[h] = __expf(s1) - __expf(s2) + LAMBDA_INIT;
  }
}

// ---------------------------------------------------------------------------
// WMMA GEMM with double-buffered LDS staging of the shared B panel.
// out[M][Nn] = A[M][K] * B[K][Nn], B supplied transposed as BT[Nn][K].
// Block = 256 threads = 8 waves; wave tile 16(M) x 128(N); block 128 x 128.
// Per k-step the 128x32-half B panel (8 KB) is staged global->LDS with async
// copies (ASYNCcnt) while the previous panel feeds the WMMAs; all 8 waves
// read operands from LDS (ds_load_b128), removing the 8x redundant global
// fetch. LDS rows padded to 40 halves (80 B = 20 banks -> conflict-free).
// MODE 0: f16 row-major out; MODE 1: f16 transposed out[n][m]; MODE 2: f32 out.
// ---------------------------------------------------------------------------
template <int MODE>
__global__ __launch_bounds__(256) void k_gemm(
    const _Float16* __restrict__ A, const _Float16* __restrict__ BT,
    _Float16* __restrict__ outH, float* __restrict__ outF,
    int M, int Nn, int K) {
  __shared__ _Float16 Bs[2][128 * 40];   // 2 x 10 KB, padded row stride 40
  const int tid = threadIdx.x;
  const int lane = tid & 31;
  const int wave = tid >> 5;
  const int lq = lane & 15;   // row (A) / col (B) within tile
  const int hs = lane >> 4;   // operand half selector
  const int m0 = blockIdx.y * 128 + wave * 16;
  const int n0 = blockIdx.x * 128;

  // Staging map: thread t -> panel row sr = t/2, 16-half chunk sc = (t&1)*16.
  const int sr = tid >> 1;
  const int sc = (tid & 1) * 16;
  const _Float16* gB = BT + (size_t)(n0 + sr) * K + sc;
  _Float16* lB0 = &Bs[0][sr * 40 + sc];
  _Float16* lB1 = &Bs[1][sr * 40 + sc];

  // Preload panel for k0 = 0 into buffer 0.
  stage_b128(gB + 0, lB0);
  stage_b128(gB + 8, lB0 + 8);

  v8f acc[8] = {};
  const _Float16* arow = A + (size_t)(m0 + lq) * K;
  for (int k0 = 0, buf = 0; k0 < K; k0 += 32, buf ^= 1) {
    WAIT_ASYNC();            // this wave's staging of Bs[buf] is complete
    __syncthreads();         // every wave's staging of Bs[buf] is complete
    if (k0 + 32 < K) {       // kick off next panel while we compute
      _Float16* lB = buf ? lB0 : lB1;
      stage_b128(gB + k0 + 32,     lB);
      stage_b128(gB + k0 + 32 + 8, lB + 8);
    }
    // A operand direct from global (per-wave rows, no cross-wave reuse).
    v16h a = load_op(arow + k0, hs);
    v16h b[8];
#pragma unroll
    for (int j = 0; j < 8; ++j)
      b[j] = load_op(&Bs[buf][(j * 16 + lq) * 40], hs);
#pragma unroll
    for (int j = 0; j < 8; ++j)
      acc[j] = wmma32(a, b[j], acc[j]);
  }
  // C layout: lane -> N = lq, VGPR v -> M = m0 + hs*8 + v
#pragma unroll
  for (int j = 0; j < 8; ++j) {
    const int n = n0 + j * 16 + lq;
#pragma unroll
    for (int v = 0; v < 8; ++v) {
      const int m = m0 + hs * 8 + v;
      if (MODE == 0)      outH[(size_t)m * Nn + n] = (_Float16)acc[j][v];
      else if (MODE == 1) outH[(size_t)n * M + m]  = (_Float16)acc[j][v];
      else                outF[(size_t)m * Nn + n] = acc[j][v];
    }
  }
}

// ---------------------------------------------------------------------------
// Fused differential flash-attention + RMS norm.
// One wave owns 16 queries of one (b,h). Computes S^T = K*Q^T so the score
// tile's C layout (lane=query, reg=key) feeds the second WMMA (O^T = V^T*P^T)
// with no cross-lane movement. Two independent online softmaxes (comp 1/2),
// O = P1 V / l1 - lam * P2 V / l2, then RMS norm over 128 dims, * 0.2.
// Loads for each 32-key step are issued as one clause before the WMMAs, and
// the next step's K/V lines are prefetched (global_prefetch_b8) while the
// current step's 24 WMMAs execute.
// ---------------------------------------------------------------------------
__global__ __launch_bounds__(128) void k_diff_attn(
    const _Float16* __restrict__ Qh, const _Float16* __restrict__ Kh,
    const _Float16* __restrict__ Vt,   // [2048 feat][4096 tokens]
    const float* __restrict__ lamBuf, const float* __restrict__ rms_scale,
    _Float16* __restrict__ Oh) {
  const int lane = threadIdx.x & 31;
  const int wid = (blockIdx.x * blockDim.x + threadIdx.x) >> 5;
  const int lq = lane & 15, hs = lane >> 4;
  const int qtiles = SEQ / 16;          // 128
  const int bh = wid / qtiles;
  const int qt = wid % qtiles;
  const int b = bh / HEADS, h = bh % HEADS;
  const int q0 = qt * 16;
  const int qglob = q0 + lq;            // this lane's query index

  const float scale = 0.125f;           // 1/sqrt(64)
  const float lam = lamBuf[h];

  // Q^T as B operand: lane = query, halves = d-contiguous. 2 comps x 2 K-chunks.
  const _Float16* qrow = Qh + (size_t)(b * SEQ + qglob) * QKV_N + h * DH2;
  v16h qop[2][2];
#pragma unroll
  for (int c = 0; c < 2; ++c)
#pragma unroll
    for (int kk = 0; kk < 2; ++kk)
      qop[c][kk] = load_op(qrow + c * DH + kk * 32, hs);

  v8f O1[8] = {}, O2[8] = {};           // O^T accumulators: 8 d-tiles x (16x16)
  float m1 = -INFINITY, m2 = -INFINITY, l1 = 0.f, l2 = 0.f;

  const _Float16* kbase = Kh + (size_t)(b * SEQ) * QKV_N + h * DH2;
  const _Float16* vhead = Vt + (size_t)(h * DH2 + lq) * TOKENS + b * SEQ;

  const int nk = q0 + 16;               // causal: keys 0..q0+15
  for (int kt = 0; kt < nk; kt += 32) { // 32 keys/step: tiles a (kt), b (kt+16)
    // ---- load clause: all 8 K operands of this step --------------------
    v16h kA[2][2], kB[2][2];
#pragma unroll
    for (int c = 0; c < 2; ++c)
#pragma unroll
      for (int kk = 0; kk < 2; ++kk) {
        const _Float16* kc = kbase + c * DH + kk * 32;
        kA[c][kk] = load_op(kc + (size_t)(kt + lq) * QKV_N, hs);
        kB[c][kk] = load_op(kc + (size_t)(kt + 16 + lq) * QKV_N, hs);
      }
    // Prefetch next step's K rows (both components) while WMMAs run.
    if (kt + 32 < nk) {
#pragma unroll
      for (int c = 0; c < 2; ++c) {
        __builtin_prefetch(kbase + c * DH + (size_t)(kt + 32 + lq) * QKV_N, 0, 1);
        __builtin_prefetch(kbase + c * DH + (size_t)(kt + 48 + lq) * QKV_N, 0, 1);
      }
    }

    // ---- score tiles (S^T) + online softmax ----------------------------
    v16h pop[2];
    float alpha[2];
#pragma unroll
    for (int c = 0; c < 2; ++c) {
      v8f sa = {}, sb = {};
#pragma unroll
      for (int kk = 0; kk < 2; ++kk) {  // accumulate over d = 64 (2 x K=32)
        sa = wmma32(kA[c][kk], qop[c][kk], sa);
        sb = wmma32(kB[c][kk], qop[c][kk], sb);
      }
      // S^T C layout: lane = query (lq), reg v -> key = kt + hs*8 + v (+16 for b)
      float pa[8], pb[8];
      float smax = -INFINITY;
#pragma unroll
      for (int v = 0; v < 8; ++v) {
        const int keyA = kt + hs * 8 + v;
        float xa = (keyA       <= qglob) ? sa[v] * scale : -INFINITY;
        float xb = (keyA + 16  <= qglob) ? sb[v] * scale : -INFINITY;
        pa[v] = xa; pb[v] = xb;
        smax = fmaxf(smax, fmaxf(xa, xb));
      }
      smax = fmaxf(smax, __shfl_xor(smax, 16, 32)); // lanes q, q+16 share a query
      const float mo = (c == 0) ? m1 : m2;
      const float mn = fmaxf(mo, smax);
      const float al = __expf(mo - mn);
      float ls = 0.f;
#pragma unroll
      for (int v = 0; v < 8; ++v) {
        pa[v] = __expf(pa[v] - mn);
        pb[v] = __expf(pb[v] - mn);
        ls += pa[v] + pb[v];
      }
      ls += __shfl_xor(ls, 16, 32);
      if (c == 0) { l1 = l1 * al + ls; m1 = mn; }
      else        { l2 = l2 * al + ls; m2 = mn; }
      alpha[c] = al;
      // Pack P^T directly as B operand (K=32 keys): halves[0..7]=tile a,
      // halves[8..15]=tile b — exactly matches the C layout we already hold.
      v16h pv;
#pragma unroll
      for (int v = 0; v < 8; ++v) { pv[v] = (_Float16)pa[v]; pv[v + 8] = (_Float16)pb[v]; }
      pop[c] = pv;
    }

    // ---- O^T += V^T * P^T ---------------------------------------------
    // Load clause: all 8 V^T operands first, then 16 WMMAs.
    const _Float16* vbase = vhead + kt;
    v16h vop[8];
#pragma unroll
    for (int t = 0; t < 8; ++t)
      vop[t] = load_op(vbase + (size_t)(t * 16) * TOKENS, hs);
    if (kt + 32 < nk) {
#pragma unroll
      for (int t = 0; t < 8; ++t)
        __builtin_prefetch(vhead + kt + 32 + (size_t)(t * 16) * TOKENS, 0, 1);
    }
#pragma unroll
    for (int t = 0; t < 8; ++t) {
#pragma unroll
      for (int v = 0; v < 8; ++v) { O1[t][v] *= alpha[0]; O2[t][v] *= alpha[1]; }
      O1[t] = wmma32(vop[t], pop[0], O1[t]);
      O2[t] = wmma32(vop[t], pop[1], O2[t]);
    }
  }

  // Combine streams, RMS-normalize over 128 dims, scale by (1 - lambda_init).
  const float inv1 = 1.0f / l1;
  const float inv2 = lam / l2;
  float ssq = 0.f;
#pragma unroll
  for (int t = 0; t < 8; ++t)
#pragma unroll
    for (int v = 0; v < 8; ++v) {
      float o = O1[t][v] * inv1 - O2[t][v] * inv2;
      ssq += o * o;
    }
  ssq += __shfl_xor(ssq, 16, 32);       // other 64 dims live in lane^16
  const float rinv = rsqrtf(ssq * (1.0f / DH2) + EPS_RMS) * (1.0f - LAMBDA_INIT);

  _Float16* orow = Oh + (size_t)(b * SEQ + qglob) * QKV_N + h * DH2;
#pragma unroll
  for (int t = 0; t < 8; ++t)
#pragma unroll
    for (int v = 0; v < 8; ++v) {
      const int d = t * 16 + hs * 8 + v;
      float o = O1[t][v] * inv1 - O2[t][v] * inv2;
      orow[d] = (_Float16)(o * rinv * rms_scale[d]);
    }
}

// ---------------------------------------------------------------------------
extern "C" void kernel_launch(void* const* d_in, const int* in_sizes, int n_in,
                              void* d_out, int out_size, void* d_ws, size_t ws_size,
                              hipStream_t stream) {
  const float* X    = (const float*)d_in[0];
  const float* Wq   = (const float*)d_in[1];
  const float* Wk   = (const float*)d_in[2];
  const float* Wv   = (const float*)d_in[3];
  const float* Wo   = (const float*)d_in[4];
  const float* lq1  = (const float*)d_in[5];
  const float* lk1  = (const float*)d_in[6];
  const float* lq2  = (const float*)d_in[7];
  const float* lk2  = (const float*)d_in[8];
  const float* rmss = (const float*)d_in[9];
  float* out = (float*)d_out;

  char* w = (char*)d_ws;
  _Float16* Xh  = (_Float16*)w; w += (size_t)TOKENS * D_MODEL * 2;  // 8 MB
  _Float16* WqT = (_Float16*)w; w += (size_t)QKV_N * D_MODEL * 2;   // 4 MB
  _Float16* WkT = (_Float16*)w; w += (size_t)QKV_N * D_MODEL * 2;
  _Float16* WvT = (_Float16*)w; w += (size_t)QKV_N * D_MODEL * 2;
  _Float16* WoT = (_Float16*)w; w += (size_t)D_MODEL * QKV_N * 2;
  _Float16* Qh  = (_Float16*)w; w += (size_t)TOKENS * QKV_N * 2;    // 16 MB
  _Float16* Kh  = (_Float16*)w; w += (size_t)TOKENS * QKV_N * 2;
  _Float16* Vt  = (_Float16*)w; w += (size_t)QKV_N * TOKENS * 2;    // transposed
  _Float16* Ohb = (_Float16*)w; w += (size_t)TOKENS * QKV_N * 2;
  float*    lam = (float*)w;

  // Stage 1: precision conversion + weight transposes + lambda scalars
  k_f32_to_f16<<<512, 256, 0, stream>>>(X, Xh, TOKENS * D_MODEL);
  k_transpose_f16<<<512, 256, 0, stream>>>(Wq, WqT, D_MODEL, QKV_N);
  k_transpose_f16<<<512, 256, 0, stream>>>(Wk, WkT, D_MODEL, QKV_N);
  k_transpose_f16<<<512, 256, 0, stream>>>(Wv, WvT, D_MODEL, QKV_N);
  k_transpose_f16<<<512, 256, 0, stream>>>(Wo, WoT, QKV_N, D_MODEL);
  k_lambda<<<1, 32, 0, stream>>>(lq1, lk1, lq2, lk2, lam);

  // Stage 2: QKV projections (WMMA). V stored transposed for the attention V^T operand.
  dim3 gq(QKV_N / 128, TOKENS / 128);
  k_gemm<0><<<gq, 256, 0, stream>>>(Xh, WqT, Qh, nullptr, TOKENS, QKV_N, D_MODEL);
  k_gemm<0><<<gq, 256, 0, stream>>>(Xh, WkT, Kh, nullptr, TOKENS, QKV_N, D_MODEL);
  k_gemm<1><<<gq, 256, 0, stream>>>(Xh, WvT, Vt, nullptr, TOKENS, QKV_N, D_MODEL);

  // Stage 3: fused differential flash attention + RMS norm
  const int waves = BATCH * HEADS * (SEQ / 16);  // 4096 waves
  k_diff_attn<<<waves / 4, 128, 0, stream>>>(Qh, Kh, Vt, lam, rmss, Ohb);

  // Stage 4: output projection -> fp32 result
  dim3 go(D_MODEL / 128, TOKENS / 128);
  k_gemm<2><<<go, 256, 0, stream>>>(Ohb, WoT, nullptr, out, TOKENS, D_MODEL, QKV_N);
}
